// MD_SSB_20529943674951
// MI455X (gfx1250) — matrix operand order repaired and checked by reading, not compiled, not run on hardware
//
#include <hip/hip_runtime.h>
#include <hip/hip_bf16.h>

// ---------------- problem constants ----------------
constexpr int NB   = 8;
constexpr int CC   = 256;
constexpr int HH   = 96;
constexpr int WW   = 96;
constexpr int HWp  = HH * WW;          // 9216
constexpr int MPOS = NB * HWp;         // 73728 positions
constexpr int HID  = 512;              // ff hidden
constexpr float EPS = 1e-6f;

typedef __attribute__((ext_vector_type(8)))  float  v8f;
typedef __attribute__((ext_vector_type(16))) __bf16 v16bf;

// ---------------- helpers ----------------
__device__ __forceinline__ unsigned short f2bf(float f) {
    unsigned u = __float_as_uint(f);
    u += 0x7fffu + ((u >> 16) & 1u);   // RNE
    return (unsigned short)(u >> 16);
}
__device__ __forceinline__ float bf2f(unsigned short h) {
    return __uint_as_float(((unsigned)h) << 16);
}
__device__ __forceinline__ float sigmoidf(float v) {
    return 1.0f / (1.0f + __expf(-v));
}

// A-fragment, 16x32 bf16, row-major A (ld = K stride).
// lanes 0-15: row M=lane, elems 0..15 = K kb+0..15
// lanes 16-31: row M=lane-16, elems 0..15 = K kb+16..31.
__device__ __forceinline__ v16bf load_frag_a_bf16(const unsigned short* __restrict__ A,
                                                  int ld, int row0, int kb, int lane) {
    int r  = row0 + (lane & 15);
    int k0 = kb + ((lane & 16) ? 16 : 0);
    const unsigned short* p = A + (size_t)r * ld + k0;
    union { v16bf v; uint4 q[2]; } u;
    u.q[0] = *reinterpret_cast<const uint4*>(p);
    u.q[1] = *reinterpret_cast<const uint4*>(p + 8);
    return u.v;
}
// B-fragment, 32x16 bf16 (K x N), weights stored row-major [N][K] (ld = K).
// lanes 0-15: col N=lane, elems 0..15 = K kb+0..15; lanes 16-31: K kb+16..31.
__device__ __forceinline__ v16bf load_frag_b_bf16(const unsigned short* __restrict__ Wt,
                                                  int ld, int n0, int kb, int lane) {
    int r  = n0 + (lane & 15);
    int k0 = kb + ((lane & 16) ? 16 : 0);
    const unsigned short* p = Wt + (size_t)r * ld + k0;
    union { v16bf v; uint4 q[2]; } u;
    u.q[0] = *reinterpret_cast<const uint4*>(p);
    u.q[1] = *reinterpret_cast<const uint4*>(p + 8);
    return u.v;
}
#define WMMA_BF16(a, b, c) \
    __builtin_amdgcn_wmma_f32_16x16x32_bf16(false, (a), false, (b), (short)0, (c), false, false)

// ---------------- small elementwise kernels ----------------
__global__ __launch_bounds__(256) void cvt_f32_bf16_kernel(const float* __restrict__ src,
                                                           unsigned short* __restrict__ dst, int n) {
    int i = blockIdx.x * 256 + threadIdx.x;
    if (i < n) dst[i] = f2bf(src[i]);
}

// LayerNorm2d statistics: one block per (n,c), reduce over H*W.
__global__ __launch_bounds__(256) void ln_stats_kernel(const float* __restrict__ X,
                                                       float* __restrict__ meanO,
                                                       float* __restrict__ rstdO) {
    const int nc = blockIdx.x;
    const float* p = X + (size_t)nc * HWp;
    float s = 0.f, s2 = 0.f;
    for (int i = threadIdx.x; i < HWp; i += 256) { float v = p[i]; s += v; s2 += v * v; }
    __shared__ float sh1[256], sh2[256];
    sh1[threadIdx.x] = s; sh2[threadIdx.x] = s2; __syncthreads();
    for (int off = 128; off > 0; off >>= 1) {
        if (threadIdx.x < off) { sh1[threadIdx.x] += sh1[threadIdx.x + off];
                                 sh2[threadIdx.x] += sh2[threadIdx.x + off]; }
        __syncthreads();
    }
    if (threadIdx.x == 0) {
        float m = sh1[0] / (float)HWp;
        float v = sh2[0] / (float)HWp - m * m;
        meanO[nc] = m; rstdO[nc] = rsqrtf(v + EPS);
    }
}

// Apply LN1 + depthwise 3x3 (+bias) -> y bf16 NHWC. Thread per NCHW element
// (coalesced 9-tap reads along w; 2B NHWC write).
__global__ __launch_bounds__(256) void ln1_dw3x3_kernel(const float* __restrict__ X,
                                                        const float* __restrict__ mean1,
                                                        const float* __restrict__ rstd1,
                                                        const float* __restrict__ ln1w,
                                                        const float* __restrict__ ln1b,
                                                        const float* __restrict__ dww,  // (3,3,1,C)
                                                        const float* __restrict__ dwb,
                                                        unsigned short* __restrict__ Y) {
    long long idx = (long long)blockIdx.x * 256 + threadIdx.x;
    if (idx >= (long long)MPOS * CC) return;
    int w  = (int)(idx % WW);
    int h  = (int)((idx / WW) % HH);
    int c  = (int)((idx / HWp) % CC);
    int n  = (int)(idx / ((long long)CC * HWp));
    const float m = mean1[n * CC + c], r = rstd1[n * CC + c];
    const float lw = ln1w[c], lb = ln1b[c];
    const float* xb = X + (((size_t)n * CC + c) * HWp);
    float acc = dwb[c];
#pragma unroll
    for (int dh = -1; dh <= 1; ++dh) {
        int hh = h + dh; if (hh < 0 || hh >= HH) continue;
#pragma unroll
        for (int dw = -1; dw <= 1; ++dw) {
            int wwp = w + dw; if (wwp < 0 || wwp >= WW) continue;
            float nv = (xb[hh * WW + wwp] - m) * r * lw + lb;
            acc += nv * dww[((dh + 1) * 3 + (dw + 1)) * CC + c];
        }
    }
    size_t mpos = (size_t)n * HWp + (size_t)h * WW + w;
    Y[mpos * CC + c] = f2bf(acc);
}

// Normalize x2 (NHWC bf16) with LN2 params -> xn bf16 NHWC.
__global__ __launch_bounds__(256) void ln2_apply_kernel(const unsigned short* __restrict__ X2,
                                                        const float* __restrict__ mean2,
                                                        const float* __restrict__ rstd2,
                                                        const float* __restrict__ ln2w,
                                                        const float* __restrict__ ln2b,
                                                        unsigned short* __restrict__ XN) {
    long long idx = (long long)blockIdx.x * 256 + threadIdx.x;
    if (idx >= (long long)MPOS * CC) return;
    int c = (int)(idx % CC);
    long long m = idx / CC;
    int nimg = (int)(m / HWp);
    float v = bf2f(X2[idx]);
    v = (v - mean2[nimg * CC + c]) * rstd2[nimg * CC + c] * ln2w[c] + ln2b[c];
    XN[idx] = f2bf(v);
}

// ---------------- WMMA GEMM kernels ----------------
// Tile: each wave computes a 32(M) x 32(N) output block: 2x2 accumulators of
// v_wmma_f32_16x16x32_bf16. 8 waves/block share the same 32 M-rows (A hits L0),
// covering the 8 N-tiles.

// Gate GEMM (N=K=256) with fused depthwise-1D conv + sigmoid-gate epilogue.
// DIR: 0=lr (axis W), 1=rl (axis W, reversed taps), 2=tb (axis H), 3=bt (axis H, reversed).
// PHASE: 0 = first (O[..] = v), 1 = middle (O[..] += v),
//        2 = last (finalize: write o as bf16 NHWC for the fuse GEMM; O f32 left dead).
template<int DIR, int PHASE>
__global__ __launch_bounds__(256) void gate_scan_kernel(const unsigned short* __restrict__ Y,
                                                        const unsigned short* __restrict__ Gw,
                                                        const float* __restrict__ Gb,
                                                        const float* __restrict__ DWw, // (5,1,C)
                                                        const float* __restrict__ DWb,
                                                        float* __restrict__ O,
                                                        unsigned short* __restrict__ OH) {
    constexpr bool AXW = (DIR < 2);
    constexpr bool REV = (DIR == 1) || (DIR == 3);
    const int lane = threadIdx.x & 31;
    const int wave = (blockIdx.x << 3) + (threadIdx.x >> 5);
    const int m0 = (wave >> 3) * 32;
    const int n0 = (wave & 7) * 32;

    const v8f vz = {0.f,0.f,0.f,0.f,0.f,0.f,0.f,0.f};
    v8f acc[2][2] = {{vz, vz}, {vz, vz}};
    for (int k = 0; k < CC; k += 32) {
        v16bf a0 = load_frag_a_bf16(Y, CC, m0,      k, lane);
        v16bf a1 = load_frag_a_bf16(Y, CC, m0 + 16, k, lane);
        v16bf b0 = load_frag_b_bf16(Gw, CC, n0,      k, lane);
        v16bf b1 = load_frag_b_bf16(Gw, CC, n0 + 16, k, lane);
        acc[0][0] = WMMA_BF16(a0, b0, acc[0][0]);
        acc[0][1] = WMMA_BF16(a0, b1, acc[0][1]);
        acc[1][0] = WMMA_BF16(a1, b0, acc[1][0]);
        acc[1][1] = WMMA_BF16(a1, b1, acc[1][1]);
    }
    const int nOff = lane & 15;
    const int mHi  = (lane & 16) ? 8 : 0;
#pragma unroll
    for (int nt = 0; nt < 2; ++nt) {
        const int d = n0 + nt * 16 + nOff;
        const float gb = Gb[d], dwb = DWb[d];
        float wt[5];
#pragma unroll
        for (int t = 0; t < 5; ++t) wt[t] = DWw[(REV ? (4 - t) : t) * CC + d];
#pragma unroll
        for (int mt = 0; mt < 2; ++mt) {
#pragma unroll
            for (int e = 0; e < 8; ++e) {
                const int m = m0 + mt * 16 + mHi + e;
                const int rem = m % HWp;
                const int h = rem / WW, w = rem % WW;
                float dwv = dwb;
#pragma unroll
                for (int t = 0; t < 5; ++t) {
                    const int p = t - 2;
                    if (AXW) {
                        int wp = w + p;
                        if (wp >= 0 && wp < WW)
                            dwv += bf2f(Y[(size_t)(m + p) * CC + d]) * wt[t];
                    } else {
                        int hp = h + p;
                        if (hp >= 0 && hp < HH)
                            dwv += bf2f(Y[(size_t)(m + p * WW) * CC + d]) * wt[t];
                    }
                }
                const float g = sigmoidf(acc[mt][nt][e] + gb);
                const float v = dwv * g;
                const size_t oi = (size_t)m * CC + d;
                if (PHASE == 0)      O[oi] = v;
                else if (PHASE == 1) O[oi] += v;
                else                 OH[oi] = f2bf(O[oi] + v);
            }
        }
    }
}

// Fuse GEMM (N=K=256): A = o (bf16 NHWC); epilogue adds residual,
// writes x2 NCHW f32 and x2 NHWC bf16.
__global__ __launch_bounds__(256) void fuse_kernel(const unsigned short* __restrict__ OH,
                                                   const unsigned short* __restrict__ Fw,
                                                   const float* __restrict__ Fb,
                                                   const float* __restrict__ X,
                                                   float* __restrict__ X2,
                                                   unsigned short* __restrict__ X2H) {
    const int lane = threadIdx.x & 31;
    const int wave = (blockIdx.x << 3) + (threadIdx.x >> 5);
    const int m0 = (wave >> 3) * 32;
    const int n0 = (wave & 7) * 32;

    const v8f vz = {0.f,0.f,0.f,0.f,0.f,0.f,0.f,0.f};
    v8f acc[2][2] = {{vz, vz}, {vz, vz}};
    for (int k = 0; k < CC; k += 32) {
        v16bf a0 = load_frag_a_bf16(OH, CC, m0,      k, lane);
        v16bf a1 = load_frag_a_bf16(OH, CC, m0 + 16, k, lane);
        v16bf b0 = load_frag_b_bf16(Fw, CC, n0,      k, lane);
        v16bf b1 = load_frag_b_bf16(Fw, CC, n0 + 16, k, lane);
        acc[0][0] = WMMA_BF16(a0, b0, acc[0][0]);
        acc[0][1] = WMMA_BF16(a0, b1, acc[0][1]);
        acc[1][0] = WMMA_BF16(a1, b0, acc[1][0]);
        acc[1][1] = WMMA_BF16(a1, b1, acc[1][1]);
    }
    const int nOff = lane & 15;
    const int mHi  = (lane & 16) ? 8 : 0;
#pragma unroll
    for (int nt = 0; nt < 2; ++nt) {
        const int d = n0 + nt * 16 + nOff;
        const float fb = Fb[d];
#pragma unroll
        for (int mt = 0; mt < 2; ++mt) {
#pragma unroll
            for (int e = 0; e < 8; ++e) {
                const int m = m0 + mt * 16 + mHi + e;
                const int nimg = m / HWp, rem = m % HWp;
                const size_t ci = ((size_t)nimg * CC + d) * HWp + rem;
                const float x2 = X[ci] + acc[mt][nt][e] + fb;
                X2[ci] = x2;
                X2H[(size_t)m * CC + d] = f2bf(x2);
            }
        }
    }
}

// ff1 GEMM (N=1024, K=256) with GLU fused: each wave computes N-tile d and
// d+512 for the same 32 M-rows, then z = a * sigmoid(g) -> bf16.
__global__ __launch_bounds__(256) void ff1_glu_kernel(const unsigned short* __restrict__ XN,
                                                      const unsigned short* __restrict__ W1,
                                                      const float* __restrict__ B1,
                                                      unsigned short* __restrict__ Z) {
    const int lane = threadIdx.x & 31;
    const int wave = (blockIdx.x << 3) + (threadIdx.x >> 5);
    const int m0 = (wave >> 4) * 32;         // 16 N-tiles per half
    const int n0 = (wave & 15) * 32;

    const v8f vz = {0.f,0.f,0.f,0.f,0.f,0.f,0.f,0.f};
    v8f acc[2][2][2];                        // [mt][nt][half]
#pragma unroll
    for (int i = 0; i < 2; ++i)
#pragma unroll
        for (int j = 0; j < 2; ++j) { acc[i][j][0] = vz; acc[i][j][1] = vz; }

    for (int k = 0; k < CC; k += 32) {
        v16bf a0  = load_frag_a_bf16(XN, CC, m0,      k, lane);
        v16bf a1  = load_frag_a_bf16(XN, CC, m0 + 16, k, lane);
        v16bf b00 = load_frag_b_bf16(W1, CC, n0,            k, lane);
        v16bf b01 = load_frag_b_bf16(W1, CC, n0 + 16,       k, lane);
        v16bf b10 = load_frag_b_bf16(W1, CC, HID + n0,      k, lane);
        v16bf b11 = load_frag_b_bf16(W1, CC, HID + n0 + 16, k, lane);
        acc[0][0][0] = WMMA_BF16(a0, b00, acc[0][0][0]);
        acc[0][1][0] = WMMA_BF16(a0, b01, acc[0][1][0]);
        acc[1][0][0] = WMMA_BF16(a1, b00, acc[1][0][0]);
        acc[1][1][0] = WMMA_BF16(a1, b01, acc[1][1][0]);
        acc[0][0][1] = WMMA_BF16(a0, b10, acc[0][0][1]);
        acc[0][1][1] = WMMA_BF16(a0, b11, acc[0][1][1]);
        acc[1][0][1] = WMMA_BF16(a1, b10, acc[1][0][1]);
        acc[1][1][1] = WMMA_BF16(a1, b11, acc[1][1][1]);
    }
    const int nOff = lane & 15;
    const int mHi  = (lane & 16) ? 8 : 0;
#pragma unroll
    for (int nt = 0; nt < 2; ++nt) {
        const int d = n0 + nt * 16 + nOff;
        const float ba = B1[d], bg = B1[d + HID];
#pragma unroll
        for (int mt = 0; mt < 2; ++mt) {
#pragma unroll
            for (int e = 0; e < 8; ++e) {
                const int m = m0 + mt * 16 + mHi + e;
                const float a = acc[mt][nt][0][e] + ba;
                const float g = acc[mt][nt][1][e] + bg;
                Z[(size_t)m * HID + d] = f2bf(a * sigmoidf(g));
            }
        }
    }
}

// ff2 GEMM (N=256, K=512) + final residual -> d_out (NCHW f32).
__global__ __launch_bounds__(256) void ff2_kernel(const unsigned short* __restrict__ Z,
                                                  const unsigned short* __restrict__ W2,
                                                  const float* __restrict__ B2,
                                                  const float* __restrict__ X2,
                                                  float* __restrict__ OUT) {
    const int lane = threadIdx.x & 31;
    const int wave = (blockIdx.x << 3) + (threadIdx.x >> 5);
    const int m0 = (wave >> 3) * 32;
    const int n0 = (wave & 7) * 32;

    const v8f vz = {0.f,0.f,0.f,0.f,0.f,0.f,0.f,0.f};
    v8f acc[2][2] = {{vz, vz}, {vz, vz}};
    for (int k = 0; k < HID; k += 32) {
        v16bf a0 = load_frag_a_bf16(Z, HID, m0,      k, lane);
        v16bf a1 = load_frag_a_bf16(Z, HID, m0 + 16, k, lane);
        v16bf b0 = load_frag_b_bf16(W2, HID, n0,      k, lane);
        v16bf b1 = load_frag_b_bf16(W2, HID, n0 + 16, k, lane);
        acc[0][0] = WMMA_BF16(a0, b0, acc[0][0]);
        acc[0][1] = WMMA_BF16(a0, b1, acc[0][1]);
        acc[1][0] = WMMA_BF16(a1, b0, acc[1][0]);
        acc[1][1] = WMMA_BF16(a1, b1, acc[1][1]);
    }
    const int nOff = lane & 15;
    const int mHi  = (lane & 16) ? 8 : 0;
#pragma unroll
    for (int nt = 0; nt < 2; ++nt) {
        const int d = n0 + nt * 16 + nOff;
        const float bb = B2[d];
#pragma unroll
        for (int mt = 0; mt < 2; ++mt) {
#pragma unroll
            for (int e = 0; e < 8; ++e) {
                const int m = m0 + mt * 16 + mHi + e;
                const int nimg = m / HWp, rem = m % HWp;
                const size_t ci = ((size_t)nimg * CC + d) * HWp + rem;
                OUT[ci] = X2[ci] + acc[mt][nt][e] + bb;
            }
        }
    }
}

// ---------------- host launcher ----------------
extern "C" void kernel_launch(void* const* d_in, const int* in_sizes, int n_in,
                              void* d_out, int out_size, void* d_ws, size_t ws_size,
                              hipStream_t stream) {
    const float* x     = (const float*)d_in[0];
    const float* ln1w  = (const float*)d_in[1];
    const float* ln1b  = (const float*)d_in[2];
    const float* dw3w  = (const float*)d_in[3];
    const float* dw3b  = (const float*)d_in[4];
    const float* dir_dww[4] = {(const float*)d_in[5],  (const float*)d_in[9],
                               (const float*)d_in[13], (const float*)d_in[17]};
    const float* dir_dwb[4] = {(const float*)d_in[6],  (const float*)d_in[10],
                               (const float*)d_in[14], (const float*)d_in[18]};
    const float* dir_gw[4]  = {(const float*)d_in[7],  (const float*)d_in[11],
                               (const float*)d_in[15], (const float*)d_in[19]};
    const float* dir_gb[4]  = {(const float*)d_in[8],  (const float*)d_in[12],
                               (const float*)d_in[16], (const float*)d_in[20]};
    const float* fusew = (const float*)d_in[21];
    const float* fuseb = (const float*)d_in[22];
    const float* ln2w  = (const float*)d_in[23];
    const float* ln2b  = (const float*)d_in[24];
    const float* ff1w  = (const float*)d_in[25];
    const float* ff1b  = (const float*)d_in[26];
    const float* ff2w  = (const float*)d_in[27];
    const float* ff2b  = (const float*)d_in[28];
    float* out = (float*)d_out;

    // workspace carve-up (all chunks 256B-aligned sizes)
    char* w = (char*)d_ws;
    size_t off = 0;
    float* mean1 = (float*)(w + off); off += (size_t)NB * CC * 4;
    float* rstd1 = (float*)(w + off); off += (size_t)NB * CC * 4;
    float* mean2 = (float*)(w + off); off += (size_t)NB * CC * 4;
    float* rstd2 = (float*)(w + off); off += (size_t)NB * CC * 4;
    unsigned short* gwb[4];
    for (int i = 0; i < 4; ++i) { gwb[i] = (unsigned short*)(w + off); off += (size_t)CC * CC * 2; }
    unsigned short* fusewb = (unsigned short*)(w + off); off += (size_t)CC * CC * 2;
    unsigned short* ff1wb  = (unsigned short*)(w + off); off += (size_t)(2 * HID) * CC * 2;
    unsigned short* ff2wb  = (unsigned short*)(w + off); off += (size_t)CC * HID * 2;
    unsigned short* ybf    = (unsigned short*)(w + off); off += (size_t)MPOS * CC * 2;
    float* obuf            = (float*)(w + off);          off += (size_t)MPOS * CC * 4;
    unsigned short* zbf    = (unsigned short*)obuf;      // reuse: o (f32) dead before z written
    float* x2nchw          = (float*)(w + off);          off += (size_t)MPOS * CC * 4;
    unsigned short* x2nhwc = (unsigned short*)(w + off); off += (size_t)MPOS * CC * 2;
    unsigned short* xnbf   = (unsigned short*)(w + off); off += (size_t)MPOS * CC * 2;
    unsigned short* ohbf   = xnbf;   // reuse: o(bf16) consumed by fuse before ln2_apply writes xn
    (void)ws_size; (void)n_in; (void)in_sizes; (void)out_size;

    // 1) weight conversion fp32 -> bf16
    auto cvt = [&](const float* s, unsigned short* d, int n) {
        cvt_f32_bf16_kernel<<<(n + 255) / 256, 256, 0, stream>>>(s, d, n);
    };
    for (int i = 0; i < 4; ++i) cvt(dir_gw[i], gwb[i], CC * CC);
    cvt(fusew, fusewb, CC * CC);
    cvt(ff1w,  ff1wb,  2 * HID * CC);
    cvt(ff2w,  ff2wb,  CC * HID);

    // 2) LN1 stats; 3) LN1 + depthwise 3x3 -> y (NHWC bf16)
    ln_stats_kernel<<<NB * CC, 256, 0, stream>>>(x, mean1, rstd1);
    ln1_dw3x3_kernel<<<(MPOS * CC) / 256, 256, 0, stream>>>(
        x, mean1, rstd1, ln1w, ln1b, dw3w, dw3b, ybf);

    // 4) four directional gate GEMMs with fused dw-conv + sigmoid epilogue;
    //    last one finalizes o directly to bf16 for the fuse GEMM.
    const int gemm_blocks = (MPOS / 32);   // 8 waves/block cover the 8 N-tiles
    gate_scan_kernel<0, 0><<<gemm_blocks, 256, 0, stream>>>(ybf, gwb[0], dir_gb[0], dir_dww[0], dir_dwb[0], obuf, nullptr);
    gate_scan_kernel<1, 1><<<gemm_blocks, 256, 0, stream>>>(ybf, gwb[1], dir_gb[1], dir_dww[1], dir_dwb[1], obuf, nullptr);
    gate_scan_kernel<2, 1><<<gemm_blocks, 256, 0, stream>>>(ybf, gwb[2], dir_gb[2], dir_dww[2], dir_dwb[2], obuf, nullptr);
    gate_scan_kernel<3, 2><<<gemm_blocks, 256, 0, stream>>>(ybf, gwb[3], dir_gb[3], dir_dww[3], dir_dwb[3], obuf, ohbf);

    // 5) fuse GEMM + residual (bf16 A operand, no in-loop conversion)
    fuse_kernel<<<gemm_blocks, 256, 0, stream>>>(ohbf, fusewb, fuseb, x, x2nchw, x2nhwc);

    // 6) LN2 stats + apply (overwrites the o-bf16 alias, which is now dead)
    ln_stats_kernel<<<NB * CC, 256, 0, stream>>>(x2nchw, mean2, rstd2);
    ln2_apply_kernel<<<(MPOS * CC) / 256, 256, 0, stream>>>(x2nhwc, mean2, rstd2, ln2w, ln2b, xnbf);

    // 7) ff1 + GLU (16 N-tiles per half -> 2x blocks), 8) ff2 + residual
    ff1_glu_kernel<<<gemm_blocks * 2, 256, 0, stream>>>(xnbf, ff1wb, ff1b, zbf);
    ff2_kernel<<<gemm_blocks, 256, 0, stream>>>(zbf, ff2wb, ff2b, x2nchw, out);
}
